// TransformerAttentionwithCatPe_34222299414555
// MI455X (gfx1250) — matrix operand order repaired and checked by reading, not compiled, not run on hardware
//
#include <hip/hip_runtime.h>
#include <math.h>

typedef __bf16 bf16_t;
typedef __attribute__((ext_vector_type(16))) bf16_t v16bf;
typedef __attribute__((ext_vector_type(8)))  bf16_t v8bf;
typedef __attribute__((ext_vector_type(16))) float  v16f;
typedef __attribute__((ext_vector_type(8)))  float  v8f;

#define B_   1024
#define S_   32
#define D_   1024
#define D2_  2048
#define XSTR 1040   // padded row stride (elements) for lds_x  (2080B: 32B-aligned, bank shift 8)
#define QSTR 528    // padded row stride (elements) for q/k chunk (1056B: 32B-aligned, bank shift 8)
#define CHUNK 512   // features per chunk (4 chunks cover 2048)

// ---------------- helpers ----------------

__device__ inline v16bf combine8(v8bf lo, v8bf hi) {
    return __builtin_shufflevector(lo, hi, 0,1,2,3,4,5,6,7,8,9,10,11,12,13,14,15);
}

// A-fragment: elements 0..7 at p, elements 8..15 at p+8 (16B-aligned each)
__device__ inline v16bf afrag16(const bf16_t* p) {
    v8bf lo = *(const v8bf*)(p);
    v8bf hi = *(const v8bf*)(p + 8);
    return combine8(lo, hi);
}

__device__ inline v8f wmma_bf16(v16bf a, v16bf b, v8f c) {
    return __builtin_amdgcn_wmma_f32_16x16x32_bf16(false, a, false, b, (short)0, c, false, false);
}

// ---------------- prep kernels (tiny vs main, run each launch) ----------------

// Wb[p][n][k] = (bf16) W_p[n*2048 + k]   for k < 1024, p in {q,k}
__global__ void prep_wb(const float* __restrict__ Wq, const float* __restrict__ Wk,
                        bf16_t* __restrict__ Wb) {
    int idx = blockIdx.x * blockDim.x + threadIdx.x;       // 2*2048*1024
    if (idx >= 2 * D2_ * D_) return;
    int p = idx >> 21;
    int r = idx & ((1 << 21) - 1);
    int n = r >> 10;
    int k = r & (D_ - 1);
    const float* W = p ? Wk : Wq;
    Wb[idx] = (bf16_t)W[n * D2_ + k];
}

// pre[p][s][n] = b_p[n] + sum_j pe[s][j] * W_p[n][1024+j]
__global__ void prep_pre(const float* __restrict__ pe,
                         const float* __restrict__ Wq, const float* __restrict__ bq,
                         const float* __restrict__ Wk, const float* __restrict__ bk,
                         float* __restrict__ pre) {
    int idx = blockIdx.x * blockDim.x + threadIdx.x;       // 2*32*2048
    if (idx >= 2 * S_ * D2_) return;
    int p = idx >> 16;
    int r = idx & 65535;
    int s = r >> 11;
    int n = r & (D2_ - 1);
    const float* W  = p ? Wk : Wq;
    const float* bb = p ? bk : bq;
    const float* pr = pe + s * D_;
    const float* wr = W + n * D2_ + D_;
    float acc = bb[n];
    for (int j = 0; j < D_; ++j) acc += pr[j] * wr[j];
    pre[idx] = acc;
}

// G[q][c] = sum_d Wfc2[q][d] * Wv[d][c]; c<1024 -> Wmv (bf16), c>=1024 -> Gp (f32)
__global__ void prep_g(const float* __restrict__ Wfc, const float* __restrict__ Wv,
                       bf16_t* __restrict__ Wmv, float* __restrict__ Gp) {
    int idx = blockIdx.x * blockDim.x + threadIdx.x;       // 32*2048
    if (idx >= S_ * D2_) return;
    int q = idx >> 11;
    int c = idx & (D2_ - 1);
    const float* fr = Wfc + q * D2_;
    float acc = 0.f;
    for (int d = 0; d < D2_; ++d) acc += fr[d] * Wv[(size_t)d * D2_ + c];
    if (c < D_) Wmv[q * D_ + c] = (bf16_t)acc;
    else        Gp[q * D_ + (c - D_)] = acc;
}

// Mpre[q][k] = sum_c pe[k][c]*Gp[q][c] + sum_d Wfc2[q][d]*bv[d]
__global__ void prep_mpre(const float* __restrict__ pe, const float* __restrict__ Wfc,
                          const float* __restrict__ bv, const float* __restrict__ Gp,
                          float* __restrict__ Mpre) {
    int idx = blockIdx.x * blockDim.x + threadIdx.x;       // 32*32
    if (idx >= S_ * S_) return;
    int q = idx >> 5;
    int k = idx & 31;
    float acc = 0.f;
    const float* pr = pe + k * D_;
    const float* gr = Gp + q * D_;
    for (int c = 0; c < D_; ++c) acc += pr[c] * gr[c];
    const float* fr = Wfc + q * D2_;
    for (int d = 0; d < D2_; ++d) acc += fr[d] * bv[d];
    Mpre[idx] = acc;
}

// ---------------- main fused kernel: one workgroup per batch ----------------

__global__ __launch_bounds__(256) void attn_main(
    const float* __restrict__ x,      // [B,32,1024]
    const bf16_t* __restrict__ Wb,    // [2][2048][1024] bf16 (B columns = W rows)
    const float* __restrict__ pre,    // [2][32][2048]
    const bf16_t* __restrict__ Wmv,   // [32][1024] bf16
    const float* __restrict__ Mpre,   // [32][32]
    const float* __restrict__ bfc,    // [1]
    float* __restrict__ out)          // [B]
{
    extern __shared__ char smem[];
    bf16_t* lds_x  = (bf16_t*)smem;                        // 32 * XSTR bf16 = 66,560 B
    bf16_t* lds_qc = lds_x + S_ * XSTR;                    // 32 * QSTR bf16 = 33,792 B
    bf16_t* lds_kc = lds_qc + S_ * QSTR;                   // 33,792 B
    float*  lds_attn = (float*)(lds_kc + S_ * QSTR);       // 32*32 f32 = 4 KB
    float*  lds_M    = lds_attn + S_ * S_;                 // 4 KB

    const int b   = blockIdx.x;
    const int tid = threadIdx.x;
    const int w   = tid >> 5;        // wave id 0..7
    const int l   = tid & 31;
    const int l15 = l & 15;
    const int lh  = l >> 4;          // lane half: 0 or 1
    const float* xb = x + (size_t)b * (S_ * D_);

    // ---- Phase 0: convert x_b -> bf16 in LDS (once per block) ----
    for (int c = tid; c < (S_ * D_) / 8; c += 256) {
        int row = c >> 7;            // 128 8-elt chunks per row
        int col = (c & 127) * 8;
        const float* sp = xb + row * D_ + col;
        float4 f0 = *(const float4*)(sp);
        float4 f1 = *(const float4*)(sp + 4);
        v8f f;
        f[0]=f0.x; f[1]=f0.y; f[2]=f0.z; f[3]=f0.w;
        f[4]=f1.x; f[5]=f1.y; f[6]=f1.z; f[7]=f1.w;
        *(v8bf*)(lds_x + row * XSTR + col) = __builtin_convertvector(f, v8bf);
    }
    __syncthreads();

    v8f attn_acc = {0.f,0.f,0.f,0.f,0.f,0.f,0.f,0.f};  // waves 0-3: attn tile
    v8f m_acc    = {0.f,0.f,0.f,0.f,0.f,0.f,0.f,0.f};  // waves 4-7: M tile

    for (int chunk = 0; chunk < 4; ++chunk) {
        // ---- Phase A: q,k projections for features [chunk*512, chunk*512+512) ----
        // (p, ntile-pair): 2 proj * 16 pairs = 32 jobs over 8 waves; each job = 2x2 tiles
        for (int j = w; j < 32; j += 8) {
            const int p   = j >> 4;               // 0=Q, 1=K
            const int np  = j & 15;               // ntile-pair within chunk
            const int ng0 = chunk * 32 + np * 2;  // global ntile of first tile

            const bf16_t* wr0 = Wb + (size_t)p * (D2_ * D_)
                                   + (size_t)(ng0 * 16 + l15) * D_ + lh * 16;
            const bf16_t* wr1 = wr0 + 16 * D_;
            const bf16_t* xr0 = lds_x + l15 * XSTR + lh * 16;          // rows 0..15
            const bf16_t* xr1 = lds_x + (16 + l15) * XSTR + lh * 16;   // rows 16..31

            v8f a00 = {0.f,0.f,0.f,0.f,0.f,0.f,0.f,0.f};
            v8f a01 = a00, a10 = a00, a11 = a00;
#pragma unroll 2
            for (int kb = 0; kb < D_; kb += 32) {
                v16bf bf0 = *(const v16bf*)(wr0 + kb);
                v16bf bf1 = *(const v16bf*)(wr1 + kb);
                v16bf af0 = afrag16(xr0 + kb);
                v16bf af1 = afrag16(xr1 + kb);
                a00 = wmma_bf16(af0, bf0, a00);
                a10 = wmma_bf16(af1, bf0, a10);
                a01 = wmma_bf16(af0, bf1, a01);
                a11 = wmma_bf16(af1, bf1, a11);
            }
            // epilogue: + (pe-part + bias), store bf16 tiles to chunk LDS
            const float* pr  = pre + (size_t)p * (S_ * D2_);
            bf16_t*      dst = p ? lds_kc : lds_qc;
            const int c0l = np * 32 + l15;        // local col within chunk (0..511)
            const int c0g = ng0 * 16 + l15;       // global feature col
#pragma unroll
            for (int i = 0; i < 8; ++i) {
                int r0 = i + lh * 8;              // rows 0..15
                int r1 = 16 + r0;                 // rows 16..31
                dst[r0 * QSTR + c0l]      = (bf16_t)(a00[i] + pr[r0 * D2_ + c0g]);
                dst[r1 * QSTR + c0l]      = (bf16_t)(a10[i] + pr[r1 * D2_ + c0g]);
                dst[r0 * QSTR + c0l + 16] = (bf16_t)(a01[i] + pr[r0 * D2_ + c0g + 16]);
                dst[r1 * QSTR + c0l + 16] = (bf16_t)(a11[i] + pr[r1 * D2_ + c0g + 16]);
            }
        }
        __syncthreads();

        // ---- Phase B: waves 0-3 accumulate attn partial; waves 4-7 compute M (chunk 0) ----
        if (w < 4) {
            const int mi = w >> 1, nj = w & 1;
            const bf16_t* qr = lds_qc + (mi * 16 + l15) * QSTR + lh * 16;
            const bf16_t* kr = lds_kc + (nj * 16 + l15) * QSTR + lh * 16;
#pragma unroll 2
            for (int d = 0; d < CHUNK; d += 32) {
                v16bf af = afrag16(qr + d);
                v16bf bf = *(const v16bf*)(kr + d);
                attn_acc = wmma_bf16(af, bf, attn_acc);
            }
        } else if (chunk == 0) {
            const int wi = w - 4, mi = wi >> 1, nj = wi & 1;
            const bf16_t* ar = Wmv + (mi * 16 + l15) * D_ + lh * 16;          // global bf16
            const bf16_t* xr = lds_x + (nj * 16 + l15) * XSTR + lh * 16;     // B col = x row
#pragma unroll 2
            for (int kb = 0; kb < D_; kb += 32) {
                v16bf af = afrag16(ar + kb);
                v16bf bf = *(const v16bf*)(xr + kb);
                m_acc = wmma_bf16(af, bf, m_acc);
            }
        }
        __syncthreads();   // protect lds_qc/kc before next chunk overwrites
    }

    // ---- Phase C: publish attn (scaled) and M (+Mpre) ----
    if (w < 4) {
        const int mi = w >> 1, nj = w & 1;
        const float scale = 0.02209708691207961f;  // 1/sqrt(2048)
#pragma unroll
        for (int i = 0; i < 8; ++i) {
            int row = mi * 16 + i + lh * 8;
            lds_attn[row * S_ + nj * 16 + l15] = attn_acc[i] * scale;
        }
    } else {
        const int wi = w - 4, mi = wi >> 1, nj = wi & 1;
#pragma unroll
        for (int i = 0; i < 8; ++i) {
            int row = mi * 16 + i + lh * 8;
            int col = nj * 16 + l15;
            lds_M[row * S_ + col] = m_acc[i] + Mpre[row * S_ + col];
        }
    }
    __syncthreads();

    // ---- Phase D: softmax over q (dim=1) per column k, dot with M, reduce, + bfc ----
    if (w == 0) {
        float mx = -3.0e38f;
#pragma unroll
        for (int q = 0; q < S_; ++q) mx = fmaxf(mx, lds_attn[q * S_ + l]);
        float den = 0.f, num = 0.f;
#pragma unroll
        for (int q = 0; q < S_; ++q) {
            float e = __expf(lds_attn[q * S_ + l] - mx);
            den += e;
            num += e * lds_M[q * S_ + l];
        }
        float yk = num / den;
#pragma unroll
        for (int off = 16; off > 0; off >>= 1)
            yk += __shfl_down(yk, off, 32);
        if (l == 0) out[b] = yk + bfc[0];
    }
}

// ---------------- launch ----------------

extern "C" void kernel_launch(void* const* d_in, const int* in_sizes, int n_in,
                              void* d_out, int out_size, void* d_ws, size_t ws_size,
                              hipStream_t stream) {
    const float* x   = (const float*)d_in[0];
    const float* pe  = (const float*)d_in[1];
    const float* Wq  = (const float*)d_in[2];
    const float* bq  = (const float*)d_in[3];
    const float* Wk  = (const float*)d_in[4];
    const float* bk  = (const float*)d_in[5];
    const float* Wv  = (const float*)d_in[6];
    const float* bv  = (const float*)d_in[7];
    const float* Wfc = (const float*)d_in[8];
    const float* bfc = (const float*)d_in[9];
    float* out = (float*)d_out;

    char* ws = (char*)d_ws;
    bf16_t* Wb   = (bf16_t*)ws;                    // 2*2048*1024*2 = 8,388,608 B
    bf16_t* Wmv  = (bf16_t*)(ws + 8388608);        // 32*1024*2    =    65,536 B
    float*  Gp   = (float*)(ws + 8454144);         // 32*1024*4    =   131,072 B
    float*  pre  = (float*)(ws + 8585216);         // 2*32*2048*4  =   524,288 B
    float*  Mpre = (float*)(ws + 9109504);         // 32*32*4      =     4,096 B

    prep_wb  <<<(2 * D2_ * D_) / 256, 256, 0, stream>>>(Wq, Wk, Wb);
    prep_pre <<<(2 * S_ * D2_) / 256, 256, 0, stream>>>(pe, Wq, bq, Wk, bk, pre);
    prep_g   <<<(S_ * D2_) / 256,     256, 0, stream>>>(Wfc, Wv, Wmv, Gp);
    prep_mpre<<<(S_ * S_ + 255) / 256,256, 0, stream>>>(pe, Wfc, bv, Gp, Mpre);

    // dynamic LDS: x(66,560) + qc(33,792) + kc(33,792) + attn(4,096) + M(4,096) = 142,336 B
    // -> 2 workgroups per WGP (284,672 B < 320 KB)
    size_t lds_bytes = (size_t)S_ * XSTR * 2 + (size_t)2 * S_ * QSTR * 2
                     + (size_t)2 * S_ * S_ * 4;
    attn_main<<<B_, 256, lds_bytes, stream>>>(x, Wb, pre, Wmv, Mpre, bfc, out);
}
